// RoformerAttention_44092134260810
// MI455X (gfx1250) — matrix operand order repaired and checked by reading, not compiled, not run on hardware
//
#include <hip/hip_runtime.h>
#include <hip/hip_bf16.h>

// ---------------------------------------------------------------------------
// Roformer attention for MI455X (gfx1250, wave32, WMMA).
// B=2, N=2048, C=1024, H=16, D=64.  bf16 WMMA with f32 accumulate.
// ---------------------------------------------------------------------------

constexpr int Bsz = 2;
constexpr int Nsq = 2048;
constexpr int Cch = 1024;
constexpr int Hh  = 16;
constexpr int Dd  = 64;
constexpr int BN  = Bsz * Nsq;

typedef __attribute__((ext_vector_type(16))) __bf16 v16bf;
typedef __attribute__((ext_vector_type(8)))  __bf16 v8bf;
typedef __attribute__((ext_vector_type(8)))  float  v8f;

#define WMMA_BF16(a, b, c) \
  __builtin_amdgcn_wmma_f32_16x16x32_bf16(false, (a), false, (b), (short)0, (c), false, false)

static __device__ inline v16bf cat8(v8bf lo, v8bf hi) {
  return __builtin_shufflevector(lo, hi, 0, 1, 2, 3, 4, 5, 6, 7, 8, 9, 10, 11, 12, 13, 14, 15);
}

// ---------------------------------------------------------------------------
// fp32 -> bf16 elementwise conversion
// ---------------------------------------------------------------------------
__global__ void cvt_f32_bf16(const float* __restrict__ src, __bf16* __restrict__ dst, int n) {
  int i = blockIdx.x * blockDim.x + threadIdx.x;
  if (i < n) dst[i] = (__bf16)src[i];
}

// ---------------------------------------------------------------------------
// QKV projection: out = x @ W^T, 16x64 output tile per wave.
// z selects Wq/Wk/Wv.  Epilogue: RoPE on Q,K; V stored transposed [b,h,d,n].
// ---------------------------------------------------------------------------
__global__ __launch_bounds__(32) void qkv_gemm(const __bf16* __restrict__ Xb,
                                               const __bf16* __restrict__ Wb,
                                               __bf16* __restrict__ Qr,
                                               __bf16* __restrict__ Kr,
                                               __bf16* __restrict__ Vt) {
  const int lane = threadIdx.x;
  const int lh   = lane & 15;
  const int hi   = lane >> 4;
  const int n64  = blockIdx.x * 64;       // column tile base (0..960)
  const int mb   = blockIdx.y * 16;       // row tile base (0..4080)
  const int z    = blockIdx.z;            // 0=Q 1=K 2=V
  const __bf16* W = Wb + (size_t)z * Cch * Cch;

  v8f acc[4] = {};

  // A operand: row m = mb+lh; contiguous k order used here (same in B): lane<16 holds K {0..15},
  // lane>=16 holds K {16..31} of each 32-chunk.
  const __bf16* xrow = Xb + (size_t)(mb + lh) * Cch + hi * 16;

  for (int kc = 0; kc < Cch / 32; ++kc) {
    v8bf alo = *(const v8bf*)(xrow);
    v8bf ahi = *(const v8bf*)(xrow + 8);
    v16bf a = cat8(alo, ahi);
#pragma unroll
    for (int nt = 0; nt < 4; ++nt) {
      // B[k][n] = W[n][k]; lane = n column, elements = contiguous k.
      const __bf16* wrow = W + (size_t)(n64 + nt * 16 + lh) * Cch + kc * 32 + hi * 16;
      v16bf b = *(const v16bf*)(wrow);
      acc[nt] = WMMA_BF16(a, b, acc[nt]);
    }
    xrow += 32;
    __builtin_prefetch(xrow + 32, 0, 1);   // global_prefetch_b8 ahead of A stream
  }

  // Epilogue: C/D layout -> lane holds col N=lh, rows M = r + 8*hi.
#pragma unroll
  for (int nt = 0; nt < 4; ++nt) {
    const int cg = n64 + nt * 16 + lh;    // global feature column
    const int hd = cg >> 6;               // head
    const int d  = cg & 63;               // within-head dim
    const int j  = d >> 1;
    // inv = 10000^(-2j/64) = exp(-j*ln(10000)/32)  (uniform over r)
    const float inv = __expf(-(float)j * (9.210340371976184f / 32.0f));
#pragma unroll
    for (int r = 0; r < 8; ++r) {
      float val = acc[nt][r];
      const int m   = mb + r + hi * 8;    // global row = b*N + token
      const int bb  = m >> 11;            // / N
      const int tok = m & (Nsq - 1);      // % N
      if (z == 2) {
        // V transposed: [b][h][d][n]
        Vt[(((size_t)bb * Hh + hd) * Dd + d) * Nsq + tok] = (__bf16)val;
      } else {
        // interleaved RoPE: pairs (2j,2j+1) live in adjacent lanes
        float part = __shfl_xor(val, 1, 32);
        float ang = (float)tok * inv;
        float sv, cv;
        __sincosf(ang, &sv, &cv);
        float res = (d & 1) ? (val * cv + part * sv) : (val * cv - part * sv);
        __bf16* dst = (z == 0) ? Qr : Kr;
        dst[(((size_t)bb * Hh + hd) * Nsq + tok) * Dd + d] = (__bf16)res;
      }
    }
  }
}

// ---------------------------------------------------------------------------
// Flash-style attention: one wave handles one (b,h) and 16 query rows,
// processing 64 key/value columns per iteration (4 S subtiles).
// 16 WMMAs per iteration vs ~64 shuffles: matrix-pipe bound.
// P is transposed from C/D layout to A layout through a 2KB LDS tile.
// ---------------------------------------------------------------------------
__global__ __launch_bounds__(32) void attn_kernel(const __bf16* __restrict__ Qr,
                                                  const __bf16* __restrict__ Kr,
                                                  const __bf16* __restrict__ Vt,
                                                  float* __restrict__ out) {
  __shared__ __attribute__((aligned(32))) __bf16 pt[16 * 64];

  const int lane = threadIdx.x;
  const int lh   = lane & 15;
  const int hi   = lane >> 4;
  const int qt   = blockIdx.x;            // query 16-row tile (0..127)
  const int bh   = blockIdx.y;            // (b*H + h), 0..31
  const int b    = bh >> 4;
  const int h    = bh & 15;

  // Q in A layout (rows = queries, K-dim = d)
  const __bf16* qrow = Qr + ((size_t)bh * Nsq + qt * 16 + lh) * Dd;
  v16bf qa[2];
#pragma unroll
  for (int c2 = 0; c2 < 2; ++c2) {
    const __bf16* p = qrow + c2 * 32 + hi * 16;
    qa[c2] = cat8(*(const v8bf*)(p), *(const v8bf*)(p + 8));
  }

  v8f o[4] = {};
  float rowm[8], rowl[8];
#pragma unroll
  for (int r = 0; r < 8; ++r) { rowm[r] = -1e30f; rowl[r] = 0.0f; }

  for (int nc = 0; nc < Nsq / 64; ++nc) {
    // ---- S = Q K^T for a 16x64 block of keys: 4 subtiles x 2 WMMAs ----
    v8f s[4] = {};
#pragma unroll
    for (int sub = 0; sub < 4; ++sub) {
      const __bf16* krow = Kr + ((size_t)bh * Nsq + nc * 64 + sub * 16 + lh) * Dd;
      v16bf kb0 = *(const v16bf*)(krow + hi * 16);        // d 0..31
      v16bf kb1 = *(const v16bf*)(krow + 32 + hi * 16);   // d 32..63
      s[sub] = WMMA_BF16(qa[0], kb0, s[sub]);
      s[sub] = WMMA_BF16(qa[1], kb1, s[sub]);
    }

    __syncthreads();  // single-wave WG: cheap; orders LDS reuse across iterations

    // ---- online softmax over 64 columns; write P (bf16) to LDS ----
#pragma unroll
    for (int r = 0; r < 8; ++r) {
      float v0 = s[0][r] * 0.125f;        // scale = D^-0.5 = 1/8
      float v1 = s[1][r] * 0.125f;
      float v2 = s[2][r] * 0.125f;
      float v3 = s[3][r] * 0.125f;
      // in-lane fold over subtiles, then one 16-lane reduction
      float mx = fmaxf(fmaxf(v0, v1), fmaxf(v2, v3));
#pragma unroll
      for (int off = 1; off < 16; off <<= 1) mx = fmaxf(mx, __shfl_xor(mx, off, 32));
      float nm = fmaxf(rowm[r], mx);
      float f  = __expf(rowm[r] - nm);
      float p0 = __expf(v0 - nm);
      float p1 = __expf(v1 - nm);
      float p2 = __expf(v2 - nm);
      float p3 = __expf(v3 - nm);
      float t = (p0 + p1) + (p2 + p3);
#pragma unroll
      for (int off = 1; off < 16; off <<= 1) t += __shfl_xor(t, off, 32);
      rowl[r] = rowl[r] * f + t;
      rowm[r] = nm;
#pragma unroll
      for (int dt = 0; dt < 4; ++dt) o[dt][r] *= f;
      const int rowbase = (r + hi * 8) * 64;
      pt[rowbase + lh]      = (__bf16)p0;
      pt[rowbase + 16 + lh] = (__bf16)p1;
      pt[rowbase + 32 + lh] = (__bf16)p2;
      pt[rowbase + 48 + lh] = (__bf16)p3;
    }

    __syncthreads();

    // ---- read P back in A layout (two 16x32 chunks), O += P V ----
    v16bf pa[2];
#pragma unroll
    for (int c2 = 0; c2 < 2; ++c2) {
      const __bf16* pp = &pt[lh * 64 + c2 * 32 + hi * 16];
      pa[c2] = cat8(*(const v8bf*)(pp), *(const v8bf*)(pp + 8));
    }
#pragma unroll
    for (int dt = 0; dt < 4; ++dt) {
      const __bf16* vrow = Vt + ((size_t)bh * Dd + dt * 16 + lh) * Nsq + nc * 64;
      v16bf vb0 = *(const v16bf*)(vrow + hi * 16);
      v16bf vb1 = *(const v16bf*)(vrow + 32 + hi * 16);
      o[dt] = WMMA_BF16(pa[0], vb0, o[dt]);
      o[dt] = WMMA_BF16(pa[1], vb1, o[dt]);
    }
  }

  // ---- normalize and write fp32 output [b][n][h*64+d] ----
#pragma unroll
  for (int dt = 0; dt < 4; ++dt) {
    const int d = dt * 16 + lh;
#pragma unroll
    for (int r = 0; r < 8; ++r) {
      const int tok = qt * 16 + r + hi * 8;
      out[((size_t)b * Nsq + tok) * Cch + h * 64 + d] = o[dt][r] / rowl[r];
    }
  }
}

// ---------------------------------------------------------------------------
// launch
// ---------------------------------------------------------------------------
extern "C" void kernel_launch(void* const* d_in, const int* in_sizes, int n_in,
                              void* d_out, int out_size, void* d_ws, size_t ws_size,
                              hipStream_t stream) {
  (void)in_sizes; (void)n_in; (void)out_size; (void)ws_size;

  const float* x  = (const float*)d_in[0];
  const float* Wq = (const float*)d_in[1];
  const float* Wk = (const float*)d_in[2];
  const float* Wv = (const float*)d_in[3];
  float* out = (float*)d_out;

  char* ws = (char*)d_ws;
  size_t off = 0;
  __bf16* Xb = (__bf16*)(ws + off); off += (size_t)BN * Cch * 2;          // 8 MB
  __bf16* Wb = (__bf16*)(ws + off); off += (size_t)3 * Cch * Cch * 2;     // 6 MB
  __bf16* Qr = (__bf16*)(ws + off); off += (size_t)Bsz * Hh * Nsq * Dd * 2;
  __bf16* Kr = (__bf16*)(ws + off); off += (size_t)Bsz * Hh * Nsq * Dd * 2;
  __bf16* Vt = (__bf16*)(ws + off); off += (size_t)Bsz * Hh * Nsq * Dd * 2;

  // fp32 -> bf16
  cvt_f32_bf16<<<(BN * Cch + 255) / 256, 256, 0, stream>>>(x, Xb, BN * Cch);
  cvt_f32_bf16<<<(Cch * Cch + 255) / 256, 256, 0, stream>>>(Wq, Wb, Cch * Cch);
  cvt_f32_bf16<<<(Cch * Cch + 255) / 256, 256, 0, stream>>>(Wk, Wb + (size_t)Cch * Cch, Cch * Cch);
  cvt_f32_bf16<<<(Cch * Cch + 255) / 256, 256, 0, stream>>>(Wv, Wb + (size_t)2 * Cch * Cch, Cch * Cch);

  // QKV projection + RoPE (grid: 16 col-tiles x 256 row-tiles x {Q,K,V})
  qkv_gemm<<<dim3(Cch / 64, BN / 16, 3), 32, 0, stream>>>(Xb, Wb, Qr, Kr, Vt);

  // attention (grid: 128 query tiles x 32 (b,h) pairs)
  attn_kernel<<<dim3(Nsq / 16, Bsz * Hh), 32, 0, stream>>>(Qr, Kr, Vt, out);
}
